// MyModel_75960791597075
// MI455X (gfx1250) — matrix operand order repaired and checked by reading, compiled, not run on hardware
//
#include <hip/hip_runtime.h>
#include <math.h>

// ---------------- model constants ----------------
constexpr int Bc   = 2;
constexpr int Sc   = 1024;
constexpr int Dc   = 1024;
constexpr int Hc   = 16;
constexpr int HDc  = 64;
constexpr int Lc   = 8;
constexpr int DFFc = 4096;
constexpr int Vc   = 32000;
constexpr float MIN_FP16 = -65504.0f;
constexpr float EPSc = 1e-12f;

// ---------------- WMMA types ----------------
typedef __bf16 v16bf __attribute__((ext_vector_type(16)));
typedef float  v8f   __attribute__((ext_vector_type(8)));

union Frag16 {
    v16bf v;
    unsigned int u[8];
};

__device__ __forceinline__ unsigned short f2bf(float f) {
    unsigned int u = __float_as_uint(f);
    u += 0x7FFFu + ((u >> 16) & 1u);   // round-to-nearest-even
    return (unsigned short)(u >> 16);
}

__device__ __forceinline__ float gelu_exact(float x) {
    return 0.5f * x * (1.0f + erff(x * 0.70710678118654752f));
}

// ---- CDNA5 async global->LDS copy (ASYNCcnt path, CDNA5 ISA) ----
// ldsDst is a pointer into static __shared__: ptrtoint both (a) yields the LDS byte
// offset (flat LDS address low 32 bits == LDS offset, ISA 10.2) and (b) CAPTURES the
// shared object so the "memory" clobber is known to possibly write it (keeps the
// compiler from folding the subsequent ds_load fragment reads to undef).
__device__ __forceinline__ void async_copy_b128(void* ldsDst, const void* gp) {
    unsigned off = (unsigned)(unsigned long long)ldsDst;
    asm volatile("global_load_async_to_lds_b128 %0, %1, off"
                 :: "v"(off), "v"(gp)
                 : "memory");
}
__device__ __forceinline__ void wait_async_zero() {
    asm volatile("s_wait_asynccnt 0" ::: "memory");
}

// ---------------- batched GEMM: C = epilogue(alpha * A@B [+ res]) ----------------
// A : bf16 [batch, M, K] row-major (lda in elems)
// B : bMode 0 -> pre-interleaved u32 words, word[kp][n] = {bf16 B[2kp,n], bf16 B[2kp+1,n]},
//                ldb = words per kp-row (= full N), sB in words
//     bMode 1 -> transB: bf16 with B[k,n] = Bp[n*ldb + k] (row-major [N,K]), sB in elems
// C : fp32 (outBf16=0) or bf16 (outBf16=1), ldc/sC in C elems. res: fp32, same geometry as C.
// Block tile 128x64x64; 8 waves; wave = 32x32 patch; 8 v_wmma_f32_16x16x32_bf16 per stage.
// Double-buffered LDS, async copies overlap WMMA issue; one barrier per K-stage.
constexpr int BM = 128;
constexpr int BN = 64;
constexpr int BK = 64;
constexpr int LDA_E   = 72;                  // padded u16 elems per A tile row (144B; 36-dword stride)
constexpr int A_BYTES = BM * LDA_E * 2;      // 18432
constexpr int LDB_E   = 72;                  // mode1 padded u16 elems per B tile row
constexpr int B_OFF   = A_BYTES;
constexpr int B_BYTES = BN * LDB_E * 2;      // 9216 (mode0 needs 32*64*4 = 8192 <= this)
constexpr int STAGE_BYTES = A_BYTES + B_BYTES;        // 27648
constexpr int LDS_BYTES   = 2 * STAGE_BYTES;          // 55296

__global__ __launch_bounds__(256) void gemm_wmma_bf16(
    const unsigned short* __restrict__ A, const void* __restrict__ Bm,
    void* __restrict__ Cv, const float* __restrict__ res,
    int M, int N, int K, int lda, int ldb, int ldc,
    long long sA, long long sB, long long sC, long long sRes,
    int bMode, float alpha, int doGelu, int outBf16)
{
    __shared__ __align__(16) unsigned char lds[LDS_BYTES];

    const int tid  = threadIdx.x;
    const int lane = tid & 31;
    const int wave = tid >> 5;
    const int lr   = lane & 15;
    const int hi   = lane >> 4;
    const int wm   = (wave & 3) * 32;
    const int wn   = (wave >> 2) * 32;

    const int bM = blockIdx.y * BM;
    const int bN = blockIdx.x * BN;
    const long long batch = blockIdx.z;

    const unsigned short* Ab = A + batch * sA;
    const unsigned int*   Bw = (const unsigned int*)Bm + batch * sB;   // mode 0
    const unsigned short* Bh = (const unsigned short*)Bm + batch * sB; // mode 1
    const float* Rb = res ? (res + batch * sRes) : nullptr;

    // stage one 128x64(A) + 64-deep(B) tile at K offset kk into buffer pb
    auto stage = [&](int kk, int pb) {
        unsigned char* base = lds + pb * STAGE_BYTES;
        // A: 1024 x 16B chunks, 4 per thread
        #pragma unroll
        for (int i = 0; i < 4; ++i) {
            int c   = tid + i * 256;
            int row = c >> 3;         // 0..127
            int kq  = c & 7;          // 0..7  (8 bf16 per chunk)
            async_copy_b128(base + row * (LDA_E * 2) + kq * 16,
                            Ab + (long long)(bM + row) * lda + kk + kq * 8);
        }
        if (bMode == 0) {
            // [kp][n] u32 words: 32 kp x 64 n -> 8192B, 2 chunks per thread
            #pragma unroll
            for (int i = 0; i < 2; ++i) {
                int c   = tid + i * 256;
                int kpl = c >> 4;          // 0..31
                int n0  = (c & 15) * 4;    // 0..60
                async_copy_b128(base + B_OFF + (kpl * BN + n0) * 4,
                                Bw + (long long)((kk >> 1) + kpl) * ldb + bN + n0);
            }
        } else {
            // [n][k] bf16 rows: 64 n x 64 k, padded stride 72 elems
            #pragma unroll
            for (int i = 0; i < 2; ++i) {
                int c  = tid + i * 256;
                int n  = c >> 3;           // 0..63
                int kq = c & 7;            // 0..7
                async_copy_b128(base + B_OFF + n * (LDB_E * 2) + kq * 16,
                                Bh + (long long)(bN + n) * ldb + kk + kq * 8);
            }
        }
    };

    v8f acc[2][2];
    #pragma unroll
    for (int i = 0; i < 2; ++i)
        #pragma unroll
        for (int j = 0; j < 2; ++j)
            #pragma unroll
            for (int r = 0; r < 8; ++r)
                acc[i][j][r] = 0.0f;

    // prologue: stage first tile
    stage(0, 0);
    wait_async_zero();
    __syncthreads();

    int p = 0;
    for (int kk = 0; kk < K; kk += BK) {
        // overlap: kick off next tile's async copies into the other buffer
        if (kk + BK < K) stage(kk + BK, p ^ 1);

        const unsigned char*  buf  = lds + p * STAGE_BYTES;
        const unsigned short* sA16 = (const unsigned short*)buf;
        const unsigned int*   sB32 = (const unsigned int*)(buf + B_OFF);
        const unsigned short* sB16 = (const unsigned short*)(buf + B_OFF);

        #pragma unroll
        for (int ks = 0; ks < 2; ++ks) {          // two 32-deep WMMA K-steps
            Frag16 bf0, bf1;
            if (bMode == 0) {
                #pragma unroll
                for (int r = 0; r < 8; ++r) {
                    bf0.u[r] = sB32[(ks * 16 + hi * 8 + r) * BN + wn +      lr];
                    bf1.u[r] = sB32[(ks * 16 + hi * 8 + r) * BN + wn + 16 + lr];
                }
            } else {
                #pragma unroll
                for (int r = 0; r < 8; ++r) {
                    bf0.u[r] = *(const unsigned int*)&sB16[(wn +      lr) * LDB_E + ks * 32 + hi * 16 + 2 * r];
                    bf1.u[r] = *(const unsigned int*)&sB16[(wn + 16 + lr) * LDB_E + ks * 32 + hi * 16 + 2 * r];
                }
            }
            #pragma unroll
            for (int sm = 0; sm < 2; ++sm) {
                Frag16 af;
                const int rowb = (wm + sm * 16 + lr) * LDA_E + ks * 32;
                #pragma unroll
                for (int r = 0; r < 4; ++r) {
                    af.u[r]     = *(const unsigned int*)&sA16[rowb + hi * 16 +     2 * r];
                    af.u[4 + r] = *(const unsigned int*)&sA16[rowb + hi * 16 + 8 + 2 * r];
                }
                acc[sm][0] = __builtin_amdgcn_wmma_f32_16x16x32_bf16(
                    false, af.v, false, bf0.v, (short)0, acc[sm][0], false, false);
                acc[sm][1] = __builtin_amdgcn_wmma_f32_16x16x32_bf16(
                    false, af.v, false, bf1.v, (short)0, acc[sm][1], false, false);
            }
        }

        wait_async_zero();     // next tile's copies done (this wave)
        __syncthreads();       // all waves done copying + consuming
        p ^= 1;
    }

    // ---- epilogue: C/D layout -> row = hi*8 + r, col = lr ----
    float*          Cf = (float*)Cv + batch * sC;
    unsigned short* Ch = (unsigned short*)Cv + batch * sC;
    #pragma unroll
    for (int sm = 0; sm < 2; ++sm) {
        #pragma unroll
        for (int sn = 0; sn < 2; ++sn) {
            const int col = bN + wn + sn * 16 + lr;
            #pragma unroll
            for (int r = 0; r < 8; ++r) {
                const int row = bM + wm + sm * 16 + hi * 8 + r;
                float val = acc[sm][sn][r] * alpha;
                if (doGelu) val = gelu_exact(val);
                if (Rb)     val += Rb[(long long)row * ldc + col];
                if (outBf16) Ch[(long long)row * ldc + col] = f2bf(val);
                else         Cf[(long long)row * ldc + col] = val;
            }
        }
    }
}

// ---------------- weight pre-swizzle: fp32 [K,N] -> bf16 pair-interleaved u32 [K/2][N] ----------------
__global__ __launch_bounds__(256) void interleave_bf16_k(const float* __restrict__ W,
                                                         unsigned short* __restrict__ out,
                                                         int N)
{
    const long long idx = (long long)blockIdx.x * 256 + threadIdx.x;  // over (K/2)*N words
    const int       n   = (int)(idx % N);
    const long long kp  = idx / N;
    out[idx * 2]     = f2bf(W[(2 * kp)     * N + n]);
    out[idx * 2 + 1] = f2bf(W[(2 * kp + 1) * N + n]);
}

// ---------------- embedding gather: h[row,:] = emb[x[row],:] (fp32) ----------------
__global__ __launch_bounds__(256) void embed_k(const int* __restrict__ x,
                                               const float* __restrict__ emb,
                                               float* __restrict__ h)
{
    const int row = blockIdx.x;
    const int t   = threadIdx.x;
    const long long tok = (long long)x[row];
    #pragma unroll
    for (int i = 0; i < Dc / 256; ++i) {
        int c = t + i * 256;
        h[(long long)row * Dc + c] = emb[tok * Dc + c];
    }
}

// ---------------- layernorm (faithful: divide by (var + eps), NO sqrt); bf16 out ----------------
__global__ __launch_bounds__(256) void layernorm_k(const float* __restrict__ in,
                                                   unsigned short* __restrict__ out,
                                                   const float* __restrict__ w,
                                                   const float* __restrict__ b)
{
    __shared__ float red[256];
    const int row = blockIdx.x;
    const int t   = threadIdx.x;
    const float* p = in + (long long)row * Dc;

    float v[Dc / 256];
    float s = 0.0f;
    #pragma unroll
    for (int i = 0; i < Dc / 256; ++i) { v[i] = p[t + i * 256]; s += v[i]; }
    red[t] = s; __syncthreads();
    for (int off = 128; off > 0; off >>= 1) {
        if (t < off) red[t] += red[t + off];
        __syncthreads();
    }
    const float mean = red[0] / (float)Dc;
    __syncthreads();

    float s2 = 0.0f;
    #pragma unroll
    for (int i = 0; i < Dc / 256; ++i) { float d = v[i] - mean; s2 += d * d; }
    red[t] = s2; __syncthreads();
    for (int off = 128; off > 0; off >>= 1) {
        if (t < off) red[t] += red[t + off];
        __syncthreads();
    }
    const float var = red[0] / (float)Dc;
    const float inv = 1.0f / (var + EPSc);   // faithful: no sqrt

    #pragma unroll
    for (int i = 0; i < Dc / 256; ++i) {
        int c = t + i * 256;
        out[(long long)row * Dc + c] = f2bf(w[c] * (v[i] - mean) * inv + b[c]);
    }
}

// ---- rope + split qkv[B,S,H,3HD](f32) -> q,k bf16 [B,H,S,HD]; v bf16 pair-interleaved over S ----
// Faithful quirk: rotary angle depends on HEAD index h, ang = h * inv[d2].
__global__ __launch_bounds__(256) void rope_split_k(const float* __restrict__ qkv,
                                                    unsigned short* __restrict__ q,
                                                    unsigned short* __restrict__ kb,
                                                    unsigned short* __restrict__ vI)
{
    const int idx = blockIdx.x * 256 + threadIdx.x;   // over B*H*S*(HD/2) = 2^20
    const int d2 = idx & 31;
    const int s  = (idx >> 5)  & (Sc - 1);
    const int hh = (idx >> 15) & (Hc - 1);
    const int b  = idx >> 19;

    const long long rowbase = ((long long)(b * Sc + s)) * (3 * Dc) + hh * (3 * HDc);
    const long long outbase = (((long long)(b * Hc + hh)) * Sc + s) * HDc + 2 * d2;

    const float e   = (float)(2 * d2) / (float)HDc;
    const float inv = powf(10000.0f, -e);
    const float ang = (float)hh * inv;
    const float cs = cosf(ang), sn = sinf(ang);

    float qa = qkv[rowbase + 2 * d2], qb = qkv[rowbase + 2 * d2 + 1];
    float ka = qkv[rowbase + HDc + 2 * d2], kcb = qkv[rowbase + HDc + 2 * d2 + 1];
    q[outbase]     = f2bf(qa * cs - qb * sn);
    q[outbase + 1] = f2bf(qa * sn + qb * cs);
    kb[outbase]     = f2bf(ka * cs - kcb * sn);
    kb[outbase + 1] = f2bf(ka * sn + kcb * cs);

    // v: pair-interleave over sequence (GEMM K dim): word[(s/2)][d] halves selected by s&1
    const long long vbase = ((long long)(b * Hc + hh)) * (Sc / 2) * HDc;  // in words
    const int sp = s >> 1, sl = s & 1;
    const int d0 = 2 * d2;
    vI[(vbase + (long long)sp * HDc + d0)     * 2 + sl] = f2bf(qkv[rowbase + 2 * HDc + d0]);
    vI[(vbase + (long long)sp * HDc + d0 + 1) * 2 + sl] = f2bf(qkv[rowbase + 2 * HDc + d0 + 1]);
}

// ---------------- causal masked softmax: scores f32 -> attn bf16 ----------------
__global__ __launch_bounds__(256) void softmax_causal_k(const float* __restrict__ sc,
                                                        unsigned short* __restrict__ attn)
{
    __shared__ float red[256];
    const int row  = blockIdx.x;           // over B*H*S
    const int qpos = row & (Sc - 1);
    const int t    = threadIdx.x;
    const float* p = sc + (long long)row * Sc;
    unsigned short* op = attn + (long long)row * Sc;

    float v[Sc / 256];
    float mx = -INFINITY;
    #pragma unroll
    for (int i = 0; i < Sc / 256; ++i) {
        int c = t + i * 256;
        float x = p[c];
        x = (c <= qpos) ? x : MIN_FP16;
        v[i] = x;
        mx = fmaxf(mx, x);
    }
    red[t] = mx; __syncthreads();
    for (int off = 128; off > 0; off >>= 1) {
        if (t < off) red[t] = fmaxf(red[t], red[t + off]);
        __syncthreads();
    }
    const float m = red[0];
    __syncthreads();

    float s = 0.0f;
    #pragma unroll
    for (int i = 0; i < Sc / 256; ++i) { v[i] = expf(v[i] - m); s += v[i]; }
    red[t] = s; __syncthreads();
    for (int off = 128; off > 0; off >>= 1) {
        if (t < off) red[t] += red[t + off];
        __syncthreads();
    }
    const float inv = 1.0f / red[0];
    #pragma unroll
    for (int i = 0; i < Sc / 256; ++i) op[t + i * 256] = f2bf(v[i] * inv);
}

// ---------------- merge heads: o[B,H,S,HD] f32 -> ot[B,S,D] bf16 ----------------
__global__ __launch_bounds__(256) void merge_heads_k(const float* __restrict__ o,
                                                     unsigned short* __restrict__ ot)
{
    const int idx = blockIdx.x * 256 + threadIdx.x;   // over B*S*D = 2^21
    const int c = idx & (Dc - 1);
    const int s = (idx >> 10) & (Sc - 1);
    const int b = idx >> 20;
    const int hh = c >> 6;
    const int d  = c & (HDc - 1);
    ot[idx] = f2bf(o[(((long long)(b * Hc + hh)) * Sc + s) * HDc + d]);
}

// ---------------- host-side launcher ----------------
static void launch_gemm(hipStream_t st,
                        const unsigned short* A, const void* Bm, void* C, const float* res,
                        int M, int N, int K, int lda, int ldb, int ldc,
                        long long sA, long long sB, long long sC, long long sRes,
                        int batch, int bMode, float alpha, int doGelu, int outBf16)
{
    dim3 g(N / BN, M / BM, batch), blk(256);
    gemm_wmma_bf16<<<g, blk, 0, st>>>(A, Bm, C, res, M, N, K, lda, ldb, ldc,
                                      sA, sB, sC, sRes, bMode, alpha, doGelu, outBf16);
}

extern "C" void kernel_launch(void* const* d_in, const int* in_sizes, int n_in,
                              void* d_out, int out_size, void* d_ws, size_t ws_size,
                              hipStream_t stream)
{
    const int*   x    = (const int*)  d_in[0];
    const float* emb  = (const float*)d_in[1];
    const float* ln1w = (const float*)d_in[2];
    const float* ln1b = (const float*)d_in[3];
    const float* Wqkv = (const float*)d_in[4];
    const float* Wo   = (const float*)d_in[5];
    const float* ln2w = (const float*)d_in[6];
    const float* ln2b = (const float*)d_in[7];
    const float* W1   = (const float*)d_in[8];
    const float* W2   = (const float*)d_in[9];
    const float* lnfw = (const float*)d_in[10];
    const float* lnfb = (const float*)d_in[11];
    const float* Wdec = (const float*)d_in[12];
    float* out = (float*)d_out;

    const long long R  = (long long)Bc * Sc;     // 2048 token rows
    const int BH = Bc * Hc;                      // 32 attention batches
    const float scale = 0.125f;                  // 1/sqrt(64)

    // ---- workspace carve (byte-based; every size is 16B-aligned) ----
    char* w = (char*)d_ws;
    auto allocF = [&](long long n) { float* p = (float*)w; w += n * 4; return p; };
    auto allocH = [&](long long n) { unsigned short* p = (unsigned short*)w; w += n * 2; return p; };

    float* h      = allocF(R * Dc);
    float* h2     = allocF(R * Dc);
    float* qkv    = allocF(R * 3 * Dc);
    float* scores = allocF((long long)BH * Sc * Sc);
    float* o      = allocF((long long)BH * Sc * HDc);
    unsigned short* xn   = allocH(R * Dc);
    unsigned short* q    = allocH((long long)BH * Sc * HDc);
    unsigned short* kbf  = allocH((long long)BH * Sc * HDc);
    unsigned short* vI   = allocH((long long)BH * Sc * HDc);
    unsigned short* attn = allocH((long long)BH * Sc * Sc);
    unsigned short* ot   = allocH(R * Dc);
    unsigned short* ff   = allocH(R * DFFc);
    unsigned short* wqkvI = allocH((long long)Lc * Dc * 3 * Dc);
    unsigned short* woI   = allocH((long long)Lc * Dc * Dc);
    unsigned short* w1I   = allocH((long long)Lc * Dc * DFFc);
    unsigned short* w2I   = allocH((long long)Lc * DFFc * Dc);
    unsigned short* wdecI = allocH((long long)Dc * Vc);

    // ---- pre-swizzle all weights to bf16 pair-interleaved (once per launch) ----
    for (int l = 0; l < Lc; ++l) {
        interleave_bf16_k<<<dim3((unsigned)((Dc / 2) * 3LL * Dc / 256)), dim3(256), 0, stream>>>(
            Wqkv + (long long)l * Dc * 3 * Dc, wqkvI + (long long)l * Dc * 3 * Dc, 3 * Dc);
        interleave_bf16_k<<<dim3((unsigned)((Dc / 2) * (long long)Dc / 256)), dim3(256), 0, stream>>>(
            Wo + (long long)l * Dc * Dc, woI + (long long)l * Dc * Dc, Dc);
        interleave_bf16_k<<<dim3((unsigned)((Dc / 2) * (long long)DFFc / 256)), dim3(256), 0, stream>>>(
            W1 + (long long)l * Dc * DFFc, w1I + (long long)l * Dc * DFFc, DFFc);
        interleave_bf16_k<<<dim3((unsigned)((DFFc / 2) * (long long)Dc / 256)), dim3(256), 0, stream>>>(
            W2 + (long long)l * DFFc * Dc, w2I + (long long)l * DFFc * Dc, Dc);
    }
    interleave_bf16_k<<<dim3((unsigned)((Dc / 2) * (long long)Vc / 256)), dim3(256), 0, stream>>>(
        Wdec, wdecI, Vc);

    // ---- forward ----
    embed_k<<<dim3((unsigned)R), dim3(256), 0, stream>>>(x, emb, h);

    for (int l = 0; l < Lc; ++l) {
        const unsigned short* wqkv = wqkvI + (long long)l * Dc * 3 * Dc;
        const unsigned short* wo   = woI   + (long long)l * Dc * Dc;
        const unsigned short* w1   = w1I   + (long long)l * Dc * DFFc;
        const unsigned short* w2   = w2I   + (long long)l * DFFc * Dc;

        // xn = LN1(h)  (bf16 out)
        layernorm_k<<<dim3((unsigned)R), dim3(256), 0, stream>>>(h, xn, ln1w + l * Dc, ln1b + l * Dc);
        // qkv = xn @ Wqkv  (fp32 out)
        launch_gemm(stream, xn, wqkv, qkv, nullptr,
                    (int)R, 3 * Dc, Dc, Dc, 3 * Dc, 3 * Dc, 0, 0, 0, 0, 1, 0, 1.0f, 0, 0);
        // rope + split (q,k plain bf16; v pair-interleaved bf16)
        rope_split_k<<<dim3((Bc * Hc * Sc * (HDc / 2)) / 256), dim3(256), 0, stream>>>(qkv, q, kbf, vI);
        // scores = (q @ k^T) / 8   [batched, transB]
        launch_gemm(stream, q, kbf, scores, nullptr,
                    Sc, Sc, HDc, HDc, HDc, Sc,
                    (long long)Sc * HDc, (long long)Sc * HDc, (long long)Sc * Sc, 0,
                    BH, 1, scale, 0, 0);
        // softmax(causal) -> attn bf16
        softmax_causal_k<<<dim3(BH * Sc), dim3(256), 0, stream>>>(scores, attn);
        // o = attn @ v   [batched, pre-interleaved B]
        launch_gemm(stream, attn, vI, o, nullptr,
                    Sc, HDc, Sc, Sc, HDc, HDc,
                    (long long)Sc * Sc, (long long)(Sc / 2) * HDc, (long long)Sc * HDc, 0,
                    BH, 0, 1.0f, 0, 0);
        // merge heads -> ot bf16
        merge_heads_k<<<dim3((unsigned)(R * Dc / 256)), dim3(256), 0, stream>>>(o, ot);
        // h2 = h + ot @ Wo  (fp32)
        launch_gemm(stream, ot, wo, h2, h,
                    (int)R, Dc, Dc, Dc, Dc, Dc, 0, 0, 0, 0, 1, 0, 1.0f, 0, 0);
        // xn = LN2(h2)
        layernorm_k<<<dim3((unsigned)R), dim3(256), 0, stream>>>(h2, xn, ln2w + l * Dc, ln2b + l * Dc);
        // ff = gelu(xn @ W1)  (bf16 out)
        launch_gemm(stream, xn, w1, ff, nullptr,
                    (int)R, DFFc, Dc, Dc, DFFc, DFFc, 0, 0, 0, 0, 1, 0, 1.0f, 1, 1);
        // h = h2 + ff @ W2  (fp32)
        launch_gemm(stream, ff, w2, h, h2,
                    (int)R, Dc, DFFc, DFFc, Dc, Dc, 0, 0, 0, 0, 1, 0, 1.0f, 0, 0);
    }

    // final LN + decoder
    layernorm_k<<<dim3((unsigned)R), dim3(256), 0, stream>>>(h, xn, lnfw, lnfb);
    launch_gemm(stream, xn, wdecI, out, nullptr,
                (int)R, Vc, Dc, Dc, Vc, Vc, 0, 0, 0, 0, 1, 0, 1.0f, 0, 0);
}
